// EX_Module_legacy_51110110822907
// MI455X (gfx1250) — matrix operand order repaired and checked by reading, not compiled, not run on hardware
//
#include <hip/hip_runtime.h>

// ---------------- types ----------------
typedef __bf16  v16bf __attribute__((ext_vector_type(16)));
typedef __bf16  v8bf  __attribute__((ext_vector_type(8)));
typedef __bf16  v4bf  __attribute__((ext_vector_type(4)));
typedef float   v8f   __attribute__((ext_vector_type(8)));
typedef float   v4f   __attribute__((ext_vector_type(4)));

// ---------------- problem constants ----------------
constexpr int B_  = 8;
constexpr int C_  = 256;
constexpr int C2_ = 128;
constexpr int HW_ = 16384;          // 128*128
constexpr int NTILES_ = HW_ / 128;  // 128 pixel-tiles per batch

// ---------------- workspace layout (float offsets, then bf16 bytes) ----------------
constexpr size_t QR_O     = 0;                                   // [B][HW]
constexpr size_t CA_O     = QR_O     + (size_t)B_*HW_;           // [B][HW]
constexpr size_t QLSUMP_O = CA_O     + (size_t)B_*HW_;           // [B][128 tiles][128]
constexpr size_t CTXP_O   = QLSUMP_O + (size_t)B_*NTILES_*C2_;   // [B][128][128]
constexpr size_t CASUMP_O = CTXP_O   + (size_t)B_*NTILES_*C2_;   // [B][128]
constexpr size_t SS_O     = CASUMP_O + (size_t)B_*NTILES_;       // [B][128][256]
constexpr size_t PP_O     = SS_O     + (size_t)B_*NTILES_*C_;    // [B][128][256]
constexpr size_t AVG_O    = PP_O     + (size_t)B_*NTILES_*C_;    // [B][128]
constexpr size_t QSTAT_O  = AVG_O    + (size_t)B_*C2_;           // [B][2] (max, 1/sum)
constexpr size_t AA_O     = QSTAT_O  + 64;                       // [B][256]
constexpr size_t POOL_O   = AA_O     + (size_t)B_*C_;            // [B][256]
constexpr size_t F32_TOT  = POOL_O   + (size_t)B_*C_;
constexpr size_t BF_BASE  = ((F32_TOT*4 + 255) & ~(size_t)255);  // byte offset, 256B aligned
constexpr size_t WP1_B    = BF_BASE;                             // w_ql  bf16 [128][256]
constexpr size_t WP2_B    = WP1_B + (size_t)C2_*C_*2;            // [w_vr;w_vl] bf16 [256][256]
constexpr size_t WP3_B    = WP2_B + (size_t)C_*C_*2;             // res_w bf16 [256][256]

#define WMMA_BF16(A,Bf,Cc) \
  __builtin_amdgcn_wmma_f32_16x16x32_bf16(false,(A),false,(Bf),(short)0,(Cc),false,false)

// sum across the 16 lanes of each half-wave (lanes 0-15 / 16-31 independently)
__device__ __forceinline__ float hsum16(float v){
  v += __shfl_xor(v, 1, 32);
  v += __shfl_xor(v, 2, 32);
  v += __shfl_xor(v, 4, 32);
  v += __shfl_xor(v, 8, 32);
  return v;
}

// A fragment (16x32 bf16, M x K) from LDS weight chunk Wl[row][32]:
// lane l<16 -> row m=l&15, K {0..15}; lane>=16 -> K {16..31}.
__device__ __forceinline__ v16bf load_a_lds(const __bf16* Wl, int row, int lane){
  const __bf16* p = Wl + row*32 + ((lane >> 4) << 4);
  v8bf lo = *(const v8bf*)(p);
  v8bf hi = *(const v8bf*)(p + 8);
  v16bf a;
#pragma unroll
  for (int i = 0; i < 8; ++i){ a[i] = lo[i]; a[8+i] = hi[i]; }
  return a;
}

// Software-pipelined MT-tile WMMA sweep: issue A-frag loads one tile ahead so
// the ds_load for m+1 overlaps the v_wmma for m (hides LDS latency).
template<int MT>
__device__ __forceinline__ void wmma_sweep(const __bf16* Wl, v16bf bf, v8f* acc, int lane){
  const int rsel = lane & 15;
  v16bf a_cur = load_a_lds(Wl, rsel, lane);
#pragma unroll
  for (int m = 0; m < MT; ++m){
    v16bf a_nxt;
    if (m + 1 < MT) a_nxt = load_a_lds(Wl, (m+1)*16 + rsel, lane);
    acc[m] = WMMA_BF16(a_cur, bf, acc[m]);
    if (m + 1 < MT) a_cur = a_nxt;
  }
}

// Stage one 32(K) x 128(N) chunk of x into LDS transposed [n][k] as bf16.
// Thread t handles a 4x4 sub-block: coalesced float4 global loads, b64 LDS stores.
__device__ __forceinline__ void stage_x_chunk(__bf16* Xl, const float* __restrict__ xg,
                                              int kc, int tid){
  const int kq = tid >> 5;      // 0..7  -> kl = kq*4+dk
  const int nq = tid & 31;      // 0..31 -> nl = nq*4+dn
  float c[4][4];
#pragma unroll
  for (int dk = 0; dk < 4; ++dk){
    v4f v = *(const v4f*)(xg + (size_t)(kc*32 + kq*4 + dk) * HW_ + nq*4);
    c[dk][0] = v[0]; c[dk][1] = v[1]; c[dk][2] = v[2]; c[dk][3] = v[3];
  }
#pragma unroll
  for (int dn = 0; dn < 4; ++dn){
    v4bf p = { (__bf16)c[0][dn], (__bf16)c[1][dn], (__bf16)c[2][dn], (__bf16)c[3][dn] };
    *(v4bf*)(Xl + (nq*4 + dn)*48 + kq*4) = p;
  }
  if (kc < 7){  // prefetch next chunk (global_prefetch_b8)
    __builtin_prefetch(xg + (size_t)((kc+1)*32 + kq*4) * HW_ + nq*4, 0, 1);
  }
}

// ---------------- init: convert weights to bf16 in workspace ----------------
__global__ __launch_bounds__(256) void k_init(const float* __restrict__ w_ql,
                                              const float* __restrict__ w_vr,
                                              const float* __restrict__ w_vl,
                                              const float* __restrict__ res_w,
                                              void* wsv){
  __bf16* wp1 = (__bf16*)((char*)wsv + WP1_B);
  __bf16* wp2 = (__bf16*)((char*)wsv + WP2_B);
  __bf16* wp3 = (__bf16*)((char*)wsv + WP3_B);
  size_t g = (size_t)blockIdx.x * 256 + threadIdx.x;
  if (g < 32768){
    wp1[g] = (__bf16)w_ql[g];
  } else if (g < 98304){
    size_t j = g - 32768; int row = (int)(j >> 8), k = (int)(j & 255);
    wp2[j] = (__bf16)(row < 128 ? w_vr[row*256 + k] : w_vl[(row-128)*256 + k]);
  } else if (g < 163840){
    size_t j = g - 98304;
    wp3[j] = (__bf16)res_w[j];
  }
}

// ---------------- pass 1: qr (VALU) + ql tile sums (WMMA) ----------------
__global__ __launch_bounds__(256) void k_pass1(const float* __restrict__ x,
                                               const float* __restrict__ w_qr,
                                               void* wsv){
  float* ws = (float*)wsv;
  const __bf16* W = (const __bf16*)((const char*)wsv + WP1_B);
  const int tid = threadIdx.x, lane = tid & 31, wid = tid >> 5;
  const int tile = blockIdx.x, b = blockIdx.y;
  const int n0 = tile * 128;
  const float* xg = x + (size_t)b * C_ * HW_ + n0;

  __shared__ alignas(32) __bf16 Xl[128 * 48];
  __shared__ alignas(16) __bf16 Wl[128 * 32];
  __shared__ float qlp[8 * 128];

  v8f acc[8] = {};
  float qacc = 0.0f;

  for (int kc = 0; kc < 8; ++kc){
    __syncthreads();
    stage_x_chunk(Xl, xg, kc, tid);
    if (tid < 128){   // stage weight chunk rows [128][32]
      const v8bf* src = (const v8bf*)(W + (size_t)tid*256 + kc*32);
      v8bf* dst = (v8bf*)(Wl + tid*32);
#pragma unroll
      for (int i = 0; i < 4; ++i) dst[i] = src[i];
    }
    __syncthreads();
    v16bf bf = *(const v16bf*)(Xl + (wid*16 + (lane & 15))*48 + ((lane >> 4) << 4));
    wmma_sweep<8>(Wl, bf, acc, lane);
    if (tid < 128){
#pragma unroll
      for (int kl = 0; kl < 32; ++kl)
        qacc += (float)Xl[tid*48 + kl] * w_qr[kc*32 + kl];
    }
  }
  if (tid < 128)
    ws[QR_O + (size_t)b*HW_ + n0 + tid] = fmaxf(qacc, 0.0f);

  const int hi = lane >> 4, nl15 = lane & 15;
#pragma unroll
  for (int m = 0; m < 8; ++m){
#pragma unroll
    for (int r = 0; r < 8; ++r){
      int co = m*16 + r + (hi << 3);
      float s = hsum16(fmaxf(acc[m][r], 0.0f));
      if (nl15 == 0) qlp[wid*128 + co] = s;
    }
  }
  __syncthreads();
  if (tid < 128){
    float s = 0.0f;
#pragma unroll
    for (int w2 = 0; w2 < 8; ++w2) s += qlp[w2*128 + tid];
    ws[QLSUMP_O + ((size_t)(b*NTILES_ + tile))*C2_ + tid] = s;
  }
}

// ---------------- stats: qr softmax stats, avg = softmax(mean relu gql) ----------------
__global__ __launch_bounds__(256) void k_stats1(void* wsv){
  float* ws = (float*)wsv;
  const int b = blockIdx.x, t = threadIdx.x;
  __shared__ float red[256];
  __shared__ float mv[128];
  const float* qr = ws + QR_O + (size_t)b*HW_;

  float lm = -1e30f;
  for (int n = t; n < HW_; n += 256) lm = fmaxf(lm, qr[n]);
  red[t] = lm; __syncthreads();
  for (int s = 128; s > 0; s >>= 1){ if (t < s) red[t] = fmaxf(red[t], red[t+s]); __syncthreads(); }
  float qmax = red[0]; __syncthreads();

  float ls = 0.0f;
  for (int n = t; n < HW_; n += 256) ls += __expf(qr[n] - qmax);
  red[t] = ls; __syncthreads();
  for (int s = 128; s > 0; s >>= 1){ if (t < s) red[t] += red[t+s]; __syncthreads(); }
  float qsum = red[0]; __syncthreads();
  if (t == 0){ ws[QSTAT_O + 2*b] = qmax; ws[QSTAT_O + 2*b + 1] = 1.0f / qsum; }

  if (t < 128){
    float s = 0.0f;
    for (int tl = 0; tl < NTILES_; ++tl)
      s += ws[QLSUMP_O + ((size_t)(b*NTILES_ + tl))*C2_ + t];
    mv[t] = s * (1.0f / (float)HW_);
  }
  __syncthreads();
  red[t] = (t < 128) ? mv[t] : -1e30f; __syncthreads();
  for (int s = 128; s > 0; s >>= 1){ if (t < s) red[t] = fmaxf(red[t], red[t+s]); __syncthreads(); }
  float m2 = red[0]; __syncthreads();
  float e = (t < 128) ? __expf(mv[t] - m2) : 0.0f;
  red[t] = e; __syncthreads();
  for (int s = 128; s > 0; s >>= 1){ if (t < s) red[t] += red[t+s]; __syncthreads(); }
  float se = red[0];
  if (t < 128) ws[AVG_O + b*C2_ + t] = e / se;
}

// ---------------- pass 2: [w_vr;w_vl] GEMM -> context partials + ca ----------------
__global__ __launch_bounds__(256) void k_pass2(const float* __restrict__ x, void* wsv){
  float* ws = (float*)wsv;
  const __bf16* W = (const __bf16*)((const char*)wsv + WP2_B);
  const int tid = threadIdx.x, lane = tid & 31, wid = tid >> 5;
  const int tile = blockIdx.x, b = blockIdx.y;
  const int n0 = tile * 128;
  const float* xg = x + (size_t)b * C_ * HW_ + n0;

  __shared__ alignas(32) __bf16 Xl[128 * 48];
  __shared__ alignas(16) __bf16 Wl[256 * 32];
  __shared__ float mask_s[128], avg_s[128];
  __shared__ float ctxp[8 * 128];
  __shared__ float caw[8];

  if (tid < 128){
    float qmax = ws[QSTAT_O + 2*b], qinv = ws[QSTAT_O + 2*b + 1];
    mask_s[tid] = __expf(ws[QR_O + (size_t)b*HW_ + n0 + tid] - qmax) * qinv;
    avg_s[tid]  = ws[AVG_O + b*C2_ + tid];
  }

  v8f acc[16] = {};
  for (int kc = 0; kc < 8; ++kc){
    __syncthreads();
    stage_x_chunk(Xl, xg, kc, tid);
    {   // stage weight chunk rows [256][32]
      const v8bf* src = (const v8bf*)(W + (size_t)tid*256 + kc*32);
      v8bf* dst = (v8bf*)(Wl + tid*32);
#pragma unroll
      for (int i = 0; i < 4; ++i) dst[i] = src[i];
    }
    __syncthreads();
    v16bf bf = *(const v16bf*)(Xl + (wid*16 + (lane & 15))*48 + ((lane >> 4) << 4));
    wmma_sweep<16>(Wl, bf, acc, lane);
  }
  const int hi = lane >> 4, nl15 = lane & 15;
  float mval = mask_s[wid*16 + nl15];
#pragma unroll
  for (int m = 0; m < 8; ++m){          // w_vr rows: context = sum_n relu*mask
#pragma unroll
    for (int r = 0; r < 8; ++r){
      int co = m*16 + r + (hi << 3);
      float s = hsum16(fmaxf(acc[m][r], 0.0f) * mval);
      if (nl15 == 0) ctxp[wid*128 + co] = s;
    }
  }
  float tacc = 0.0f;                    // w_vl rows: theta -> ca
#pragma unroll
  for (int m = 8; m < 16; ++m){
#pragma unroll
    for (int r = 0; r < 8; ++r){
      int c2 = (m-8)*16 + r + (hi << 3);
      tacc += avg_s[c2] * fmaxf(acc[m][r], 0.0f);
    }
  }
  tacc += __shfl_xor(tacc, 16, 32);
  float ca = 1.0f / (1.0f + __expf(-tacc));
  if (lane < 16) ws[CA_O + (size_t)b*HW_ + n0 + wid*16 + lane] = ca;
  float cs = hsum16(ca);
  if (lane == 0) caw[wid] = cs;
  __syncthreads();
  if (tid < 128){
    float s = 0.0f;
#pragma unroll
    for (int w2 = 0; w2 < 8; ++w2) s += ctxp[w2*128 + tid];
    ws[CTXP_O + ((size_t)(b*NTILES_ + tile))*C2_ + tid] = s;
  }
  if (tid == 0){
    float s = 0.0f;
    for (int w2 = 0; w2 < 8; ++w2) s += caw[w2];
    ws[CASUMP_O + b*NTILES_ + tile] = s;
  }
}

// ---------------- mid: LN -> spatial -> SK gate -> A ----------------
__global__ __launch_bounds__(256) void k_mid(const float* __restrict__ w_up,
                                             const float* __restrict__ sk_w1,
                                             const float* __restrict__ sk_w2,
                                             void* wsv){
  float* ws = (float*)wsv;
  const int b = blockIdx.x, t = threadIdx.x;
  __shared__ float ctx_s[128], sp_s[128], sa_s[256], u_s[256], u1_s[128], red[256];

  if (t < 128){
    float s = 0.0f;
    for (int tl = 0; tl < NTILES_; ++tl)
      s += ws[CTXP_O + ((size_t)(b*NTILES_ + tl))*C2_ + t];
    ctx_s[t] = s;
  }
  __syncthreads();
  red[t] = (t < 128) ? ctx_s[t] : 0.0f; __syncthreads();
  for (int s = 128; s > 0; s >>= 1){ if (t < s) red[t] += red[t+s]; __syncthreads(); }
  float mu = red[0] * (1.0f/128.0f); __syncthreads();
  red[t] = (t < 128) ? (ctx_s[t]-mu)*(ctx_s[t]-mu) : 0.0f; __syncthreads();
  for (int s = 128; s > 0; s >>= 1){ if (t < s) red[t] += red[t+s]; __syncthreads(); }
  float var = red[0] * (1.0f/128.0f); __syncthreads();
  if (t < 128){
    float ln = (ctx_s[t]-mu) * rsqrtf(var + 1e-5f);
    sp_s[t] = 1.0f / (1.0f + __expf(-ln));
  }
  red[t] = (t < 128) ? ws[CASUMP_O + b*NTILES_ + t] : 0.0f; __syncthreads();
  for (int s = 128; s > 0; s >>= 1){ if (t < s) red[t] += red[t+s]; __syncthreads(); }
  float mca = red[0] * (1.0f / (float)HW_); __syncthreads();

  {
    float d = 0.0f;
    for (int j = 0; j < 128; ++j) d += w_up[t*128 + j] * sp_s[j];
    float sa = fminf(fmaxf((d + 3.0f) * (1.0f/6.0f), 0.0f), 1.0f);
    sa_s[t] = sa;
    u_s[t]  = sa * (1.0f + mca) + mca;
  }
  __syncthreads();
  if (t < 128){
    float d = 0.0f;
    for (int j = 0; j < 256; ++j) d += sk_w1[t*256 + j] * u_s[j];
    u1_s[t] = fmaxf(d, 0.0f);
  }
  __syncthreads();
  float u2;
  {
    float d = 0.0f;
    for (int j = 0; j < 128; ++j) d += sk_w2[t*128 + j] * u1_s[j];
    u2 = fmaxf(d, 0.0f);
  }
  red[t] = u2; __syncthreads();
  for (int s = 128; s > 0; s >>= 1){ if (t < s) red[t] = fmaxf(red[t], red[t+s]); __syncthreads(); }
  float m = red[0]; __syncthreads();
  float e = __expf(u2 - m);
  red[t] = e; __syncthreads();
  for (int s = 128; s > 0; s >>= 1){ if (t < s) red[t] += red[t+s]; __syncthreads(); }
  float a = e / red[0];
  ws[AA_O + b*C_ + t] = 1.0f - a * (1.0f - sa_s[t]);   // A = a*sa + (1-a)
}

// ---------------- pass 3: res_w GEMM fused with exp(A*ca) S/P partials ----------------
__global__ __launch_bounds__(256) void k_pass3(const float* __restrict__ x, void* wsv){
  float* ws = (float*)wsv;
  const __bf16* W = (const __bf16*)((const char*)wsv + WP3_B);
  const int tid = threadIdx.x, lane = tid & 31, wid = tid >> 5;
  const int tile = blockIdx.x, b = blockIdx.y;
  const int n0 = tile * 128;
  const float* xg = x + (size_t)b * C_ * HW_ + n0;

  __shared__ alignas(32) __bf16 Xl[128 * 48];
  __shared__ alignas(16) __bf16 Wl[256 * 32];
  __shared__ float ca_s[128], A_s[256];
  __shared__ float Sp[8 * 256], Pp[8 * 256];

  if (tid < 128) ca_s[tid] = ws[CA_O + (size_t)b*HW_ + n0 + tid];
  A_s[tid] = ws[AA_O + b*C_ + tid];

  v8f acc[16] = {};
  for (int kc = 0; kc < 8; ++kc){
    __syncthreads();
    stage_x_chunk(Xl, xg, kc, tid);
    {
      const v8bf* src = (const v8bf*)(W + (size_t)tid*256 + kc*32);
      v8bf* dst = (v8bf*)(Wl + tid*32);
#pragma unroll
      for (int i = 0; i < 4; ++i) dst[i] = src[i];
    }
    __syncthreads();
    v16bf bf = *(const v16bf*)(Xl + (wid*16 + (lane & 15))*48 + ((lane >> 4) << 4));
    wmma_sweep<16>(Wl, bf, acc, lane);
  }
  const int hi = lane >> 4, nl15 = lane & 15;
  float ca_l = ca_s[wid*16 + nl15];
#pragma unroll
  for (int m = 0; m < 16; ++m){
#pragma unroll
    for (int r = 0; r < 8; ++r){
      int co = m*16 + r + (hi << 3);
      float y = fmaxf(acc[m][r], 0.0f);
      float e = __expf(A_s[co] * ca_l);     // A*ca in (0,1): no max-sub needed
      float se = hsum16(e);
      float pe = hsum16(e * y);
      if (nl15 == 0){ Sp[wid*256 + co] = se; Pp[wid*256 + co] = pe; }
    }
  }
  __syncthreads();
  {
    float sS = 0.0f, sP = 0.0f;
#pragma unroll
    for (int w2 = 0; w2 < 8; ++w2){ sS += Sp[w2*256 + tid]; sP += Pp[w2*256 + tid]; }
    size_t o = ((size_t)(b*NTILES_ + tile))*C_ + tid;
    ws[SS_O + o] = sS;
    ws[PP_O + o] = sP;
  }
}

// ---------------- pooled = P/S ----------------
__global__ __launch_bounds__(256) void k_pool(void* wsv){
  float* ws = (float*)wsv;
  const int b = blockIdx.x, t = threadIdx.x;
  float sS = 0.0f, sP = 0.0f;
  for (int tl = 0; tl < NTILES_; ++tl){
    size_t o = ((size_t)(b*NTILES_ + tl))*C_ + t;
    sS += ws[SS_O + o];
    sP += ws[PP_O + o];
  }
  ws[POOL_O + b*C_ + t] = sP / sS;
}

// ---------------- out = x + pooled (NT load/store: out is write-once) ----------------
__global__ __launch_bounds__(256) void k_final(const float* __restrict__ x,
                                               const void* wsv,
                                               float* __restrict__ out){
  const float* ws = (const float*)wsv;
  size_t i = ((size_t)blockIdx.x * 256 + threadIdx.x) * 4;
  int bc = (int)(i >> 14);                  // HW = 2^14
  float p = ws[POOL_O + bc];
  v4f xv = __builtin_nontemporal_load((const v4f*)(x + i));
  v4f o; o[0] = xv[0] + p; o[1] = xv[1] + p; o[2] = xv[2] + p; o[3] = xv[3] + p;
  __builtin_nontemporal_store(o, (v4f*)(out + i));
}

// ---------------- launcher ----------------
extern "C" void kernel_launch(void* const* d_in, const int* in_sizes, int n_in,
                              void* d_out, int out_size, void* d_ws, size_t ws_size,
                              hipStream_t stream) {
  const float* x     = (const float*)d_in[0];
  const float* w_qr  = (const float*)d_in[1];
  const float* w_vr  = (const float*)d_in[2];
  const float* w_up  = (const float*)d_in[3];
  const float* w_ql  = (const float*)d_in[4];
  const float* w_vl  = (const float*)d_in[5];
  const float* sk_w1 = (const float*)d_in[6];
  const float* sk_w2 = (const float*)d_in[7];
  const float* res_w = (const float*)d_in[8];
  float* out = (float*)d_out;

  dim3 gp(NTILES_, B_);   // 128 pixel-tiles x 8 batches

  k_init  <<<640, 256, 0, stream>>>(w_ql, w_vr, w_vl, res_w, d_ws);
  k_pass1 <<<gp,  256, 0, stream>>>(x, w_qr, d_ws);
  k_stats1<<<B_,  256, 0, stream>>>(d_ws);
  k_pass2 <<<gp,  256, 0, stream>>>(x, d_ws);
  k_mid   <<<B_,  256, 0, stream>>>(w_up, sk_w1, sk_w2, d_ws);
  k_pass3 <<<gp,  256, 0, stream>>>(x, d_ws);
  k_pool  <<<B_,  256, 0, stream>>>(d_ws);
  k_final <<<(B_*C_*HW_)/4/256, 256, 0, stream>>>(x, d_ws, out);
}
